// KANtoMLP_19859928776983
// MI455X (gfx1250) — matrix operand order repaired and checked by reading, not compiled, not run on hardware
//
#include <hip/hip_runtime.h>
#include <hip/hip_bf16.h>
#include <math.h>

// ---------------------------------------------------------------------------
// KAN layer fused as one bf16 GEMM on gfx1250 WMMA.
//   K-feature layout per input channel h (9 features):
//     j=0   : silu(x[n,h])                <->  base_weight[d,h]
//     j=1..8: B-spline bases_k(x[n,h])    <->  spline_weight[d,h,k]*scaler[d,h]
//   out = GELU_exact( A(8192x9216) @ W^T(9216x4096) )
// Pipeline: async global->LDS copies (ASYNCcnt) + double-buffered LDS +
//           v_wmma_f32_16x16x32_bf16 accumulation.
// ---------------------------------------------------------------------------

typedef __attribute__((ext_vector_type(16))) __bf16 v16bf;
typedef __attribute__((ext_vector_type(8)))  float  v8f;
typedef int b128_t __attribute__((vector_size(16)));   // matches builtin param type

static constexpr int H    = 1024;       // input channels
static constexpr int NB   = 8;          // spline basis count (grid 5 + order 3)
static constexpr int KPH  = 9;          // features per channel (1 silu + 8 bases)
static constexpr int K    = H * KPH;    // 9216 fused reduction dim
static constexpr int BM   = 128;        // block tile M
static constexpr int BN   = 128;        // block tile N
static constexpr int BK   = 64;         // K step (= two bf16 WMMA K-steps)
static constexpr int LDSP = 72;         // padded LDS row (ushorts): 144B rows, 16B aligned,
                                        // 36-dword bank stride -> conflict-free frag reads

#if defined(__HIP_DEVICE_COMPILE__) && __has_builtin(__builtin_amdgcn_global_load_async_to_lds_b128)
#define HAVE_ASYNC_LDS 1
#else
#define HAVE_ASYNC_LDS 0
#endif

#if defined(__HIP_DEVICE_COMPILE__) && __has_builtin(__builtin_amdgcn_s_wait_asynccnt)
#define ASYNC_WAIT() __builtin_amdgcn_s_wait_asynccnt(0)
#elif HAVE_ASYNC_LDS
#define ASYNC_WAIT() asm volatile("s_wait_asynccnt 0x0" ::: "memory")
#else
#define ASYNC_WAIT()
#endif

__device__ __forceinline__ unsigned short f2bf(float f) {
  // round-to-nearest-even fp32 -> bf16
  unsigned int u = __float_as_uint(f);
  u += 0x7FFFu + ((u >> 16) & 1u);
  return (unsigned short)(u >> 16);
}

// ---------------------------------------------------------------------------
// Stage 1a: activations -> bf16 A matrix [Nrows][K], K-contiguous.
// One thread per (n,h): silu + Cox-de Boor cubic recursion on uniform grid
// g[i] = -2.2 + 0.4*i  (i = 0..11), exactly matching the reference.
// ---------------------------------------------------------------------------
__global__ __launch_bounds__(256) void kan_prep_a(const float* __restrict__ x,
                                                  unsigned short* __restrict__ A,
                                                  int total) {
  const int idx = blockIdx.x * 256 + threadIdx.x;
  if (idx >= total) return;
  const float xv = x[idx];

  const float s = xv / (1.0f + __expf(-xv));   // SiLU

  const float g0 = -2.2f, gh = 0.4f;
  float b0[11];
#pragma unroll
  for (int t = 0; t < 11; ++t) {
    const float gl = g0 + gh * (float)t;
    b0[t] = (xv >= gl && xv < gl + gh) ? 1.0f : 0.0f;
  }
  float b1[10];
  {
    const float inv = 1.0f / gh;
#pragma unroll
    for (int t = 0; t < 10; ++t) {
      const float gt = g0 + gh * (float)t;
      b1[t] = (xv - gt) * inv * b0[t] + ((gt + 2.0f * gh) - xv) * inv * b0[t + 1];
    }
  }
  float b2[9];
  {
    const float inv = 1.0f / (2.0f * gh);
#pragma unroll
    for (int t = 0; t < 9; ++t) {
      const float gt = g0 + gh * (float)t;
      b2[t] = (xv - gt) * inv * b1[t] + ((gt + 3.0f * gh) - xv) * inv * b1[t + 1];
    }
  }
  float b3[8];
  {
    const float inv = 1.0f / (3.0f * gh);
#pragma unroll
    for (int t = 0; t < 8; ++t) {
      const float gt = g0 + gh * (float)t;
      b3[t] = (xv - gt) * inv * b2[t] + ((gt + 4.0f * gh) - xv) * inv * b2[t + 1];
    }
  }

  unsigned short* dst = A + (size_t)idx * KPH;   // == n*K + h*KPH
  dst[0] = f2bf(s);
#pragma unroll
  for (int j = 0; j < NB; ++j) dst[1 + j] = f2bf(b3[j]);
}

// ---------------------------------------------------------------------------
// Stage 1b: fused weights -> bf16 W matrix [D][K], K-contiguous.
// ---------------------------------------------------------------------------
__global__ __launch_bounds__(256) void kan_prep_w(const float* __restrict__ bw,
                                                  const float* __restrict__ sw,
                                                  const float* __restrict__ sc,
                                                  unsigned short* __restrict__ W,
                                                  int total) {
  const int idx = blockIdx.x * 256 + threadIdx.x;   // d*H + h
  if (idx >= total) return;
  const float scale = sc[idx];
  unsigned short* dst = W + (size_t)idx * KPH;      // == d*K + h*KPH
  dst[0] = f2bf(bw[idx]);
  const float* swp = sw + (size_t)idx * NB;
#pragma unroll
  for (int j = 0; j < NB; ++j) dst[1 + j] = f2bf(swp[j] * scale);
}

// ---------------------------------------------------------------------------
// Stage 2: bf16 GEMM + exact GELU epilogue.
// 256 threads = 8 wave32. Block tile 128x128, K step 64 (2 WMMA K-steps).
// Wave grid 4(M) x 2(N): each wave owns 32x64 = 2x4 WMMA tiles.
// Async global->LDS double-buffered pipeline when available.
// ---------------------------------------------------------------------------
__global__ __launch_bounds__(256) void kan_gemm(const unsigned short* __restrict__ A,
                                                const unsigned short* __restrict__ W,
                                                float* __restrict__ out,
                                                int Nrows, int D) {
#if HAVE_ASYNC_LDS
  __shared__ unsigned short lds_a[2][BM][LDSP];
  __shared__ unsigned short lds_b[2][BN][LDSP];
#else
  __shared__ unsigned short lds_a[1][BM][LDSP];
  __shared__ unsigned short lds_b[1][BN][LDSP];
#endif

  const int tid  = threadIdx.x;
  const int lane = tid & 31;
  const int wid  = tid >> 5;
  const int wm   = wid & 3;     // 0..3 -> 32-row strip
  const int wn   = wid >> 2;    // 0..1 -> 64-col strip
  const int lh   = lane & 15;
  const int half = lane >> 4;

  const int tm0 = blockIdx.x * BM;
  const int tn0 = blockIdx.y * BN;

  // global->LDS staging: 2 threads per row; each thread moves 4x b128 (32 bf16)
  // per tile per matrix -> full 128x64 bf16 tile coverage with 256 threads.
  const int ldr = tid >> 1;            // 0..127
  const int ldc = (tid & 1) << 5;      // 0 or 32 (ushort units)

  const unsigned short* ga = A + (size_t)(tm0 + ldr) * K + ldc;
  const unsigned short* gb = W + (size_t)(tn0 + ldr) * K + ldc;

  const v8f vzero = {0.f, 0.f, 0.f, 0.f, 0.f, 0.f, 0.f, 0.f};
  v8f acc[2][4];
#pragma unroll
  for (int i = 0; i < 2; ++i)
#pragma unroll
    for (int j = 0; j < 4; ++j) acc[i][j] = vzero;

  // ---- fragment build + 8 WMMAs per 32-wide K-step, over buffer `buf` ----
  auto compute = [&](int buf) {
#pragma unroll
    for (int kk = 0; kk < BK; kk += 32) {
      // A fragment (16x32 bf16): lane m=lh; half0 -> K 0..15; half1 -> K 16..31
      v16bf afr[2];
#pragma unroll
      for (int tm = 0; tm < 2; ++tm) {
        union { unsigned int u[8]; v16bf v; } f;
        const int m = wm * 32 + tm * 16 + lh;
#pragma unroll
        for (int r = 0; r < 8; ++r) {
          const int ko = 16 * half + 2 * r;
          f.u[r] = *(const unsigned int*)&lds_a[buf][m][kk + ko];
        }
        afr[tm] = f.v;
      }
      // B fragment (32x16 bf16): lane n=lh; half0 -> K 0..15; half1 -> K 16..31
      v16bf bfr[4];
#pragma unroll
      for (int tn = 0; tn < 4; ++tn) {
        union { unsigned int u[8]; v16bf v; } f;
        const int n = wn * 64 + tn * 16 + lh;
#pragma unroll
        for (int r = 0; r < 8; ++r)
          f.u[r] = *(const unsigned int*)&lds_b[buf][n][kk + 16 * half + 2 * r];
        bfr[tn] = f.v;
      }
#pragma unroll
      for (int tm = 0; tm < 2; ++tm)
#pragma unroll
        for (int tn = 0; tn < 4; ++tn)
          acc[tm][tn] = __builtin_amdgcn_wmma_f32_16x16x32_bf16(
              false, afr[tm], false, bfr[tn], (short)0, acc[tm][tn], false, false);
    }
  };

#if HAVE_ASYNC_LDS
  // ---- async (ASYNCcnt) global -> LDS copy of one 128x64 A tile + B tile ----
  auto copy_tiles = [&](int buf, int k0) {
#pragma unroll
    for (int c = 0; c < 4; ++c) {
      __builtin_amdgcn_global_load_async_to_lds_b128(
          (__attribute__((address_space(1))) b128_t*)(ga + k0 + c * 8),
          (__attribute__((address_space(3))) b128_t*)&lds_a[buf][ldr][ldc + c * 8],
          0, 0);
      __builtin_amdgcn_global_load_async_to_lds_b128(
          (__attribute__((address_space(1))) b128_t*)(gb + k0 + c * 8),
          (__attribute__((address_space(3))) b128_t*)&lds_b[buf][ldr][ldc + c * 8],
          0, 0);
    }
  };

  copy_tiles(0, 0);
  ASYNC_WAIT();
  __syncthreads();

  for (int k0 = 0; k0 < K; k0 += BK) {
    const int cur = (k0 >> 6) & 1;
    if (k0 + BK < K) copy_tiles(cur ^ 1, k0 + BK);   // prefetch next tile into other buffer
    compute(cur);
    ASYNC_WAIT();                                    // next tile resident in LDS
    __syncthreads();                                 // all waves done with `cur`, see next tile
  }
#else
  // ---- fallback: VGPR-staged copy, single buffer ----
  for (int k0 = 0; k0 < K; k0 += BK) {
    uint4 va[4], vb[4];
#pragma unroll
    for (int c = 0; c < 4; ++c) {
      va[c] = *(const uint4*)(ga + k0 + c * 8);
      vb[c] = *(const uint4*)(gb + k0 + c * 8);
    }
    if (k0 + BK < K) {
      __builtin_prefetch(ga + k0 + BK, 0, 3);
      __builtin_prefetch(gb + k0 + BK, 0, 3);
    }
    __syncthreads();      // previous iteration's readers are done
#pragma unroll
    for (int c = 0; c < 4; ++c) {
      *(uint4*)&lds_a[0][ldr][ldc + c * 8] = va[c];
      *(uint4*)&lds_b[0][ldr][ldc + c * 8] = vb[c];
    }
    __syncthreads();
    compute(0);
  }
#endif

  // Epilogue: C/D layout lane n=lh, row m = r + 8*half. Exact GELU.
  const int gm_base = tm0 + wm * 32;
  const int gn_base = tn0 + wn * 64;
#pragma unroll
  for (int tm = 0; tm < 2; ++tm) {
#pragma unroll
    for (int tn = 0; tn < 4; ++tn) {
      const int gn = gn_base + tn * 16 + lh;
#pragma unroll
      for (int r = 0; r < 8; ++r) {
        const int gm = gm_base + tm * 16 + r + 8 * half;
        const float v = acc[tm][tn][r];
        out[(size_t)gm * D + gn] = 0.5f * v * (1.0f + erff(v * 0.70710678118f));
      }
    }
  }
}

// ---------------------------------------------------------------------------
// Launch: prep A (151 MB ws), prep W (75.5 MB ws), then tiled WMMA GEMM.
// ---------------------------------------------------------------------------
extern "C" void kernel_launch(void* const* d_in, const int* in_sizes, int n_in,
                              void* d_out, int out_size, void* d_ws, size_t ws_size,
                              hipStream_t stream) {
  const float* x  = (const float*)d_in[0];   // (B,S,H)
  const float* bw = (const float*)d_in[1];   // (D,H)
  const float* sw = (const float*)d_in[2];   // (D,H,8)
  const float* sc = (const float*)d_in[3];   // (D,H)
  float* out = (float*)d_out;                // (B,S,D)

  const int Nrows = in_sizes[0] / H;         // 8192
  const int D     = in_sizes[1] / H;         // 4096

  unsigned short* Abuf = (unsigned short*)d_ws;
  unsigned short* Wbuf = Abuf + (size_t)Nrows * K;

  const int totA = Nrows * H;
  kan_prep_a<<<(totA + 255) / 256, 256, 0, stream>>>(x, Abuf, totA);
  const int totW = D * H;
  kan_prep_w<<<(totW + 255) / 256, 256, 0, stream>>>(bw, sw, sc, Wbuf, totW);

  dim3 grid(Nrows / BM, D / BN);
  kan_gemm<<<grid, 256, 0, stream>>>(Abuf, Wbuf, out, Nrows, D);
}
